// Block_transformer_cat_78520592105820
// MI455X (gfx1250) — matrix operand order repaired and checked by reading, not compiled, not run on hardware
//
#include <hip/hip_runtime.h>

#define DEV __device__ __forceinline__

typedef __attribute__((ext_vector_type(16))) _Float16     v16h;
typedef __attribute__((ext_vector_type(8)))  float        v8f;
typedef __attribute__((ext_vector_type(4)))  unsigned int v4u;
typedef __attribute__((ext_vector_type(2)))  unsigned int v2u;

union V16H { v16h h; v4u u[2]; };
union V8F  { v8f  v; float f[8]; };
union H4   { _Float16 h[4]; v2u u; };

// Load a 16-half (32B) WMMA fragment as two 16B chunks (ds_load_b128 / global_load_b128).
DEV v16h frag16(const _Float16* p) {
  V16H r;
  r.u[0] = *(const v4u*)(p);
  r.u[1] = *(const v4u*)(p + 16);
  return r.h;
}

DEV v8f wmma_f16(v16h a, v16h b, v8f c) {
  // D = A(16x32 f16) x B(32x16 f16) + C(16x16 f32)
  return __builtin_amdgcn_wmma_f32_16x16x32_f16(false, a, false, b, (short)0, c, false, false);
}

// DPP16 lane swap (VALU-only, no LDS pipe). CTRL: 0xB1=quad_perm[1,0,3,2] (xor1),
// 0x4E=quad_perm[2,3,0,1] (xor2), 0x141=row_half_mirror, 0x140=row_mirror.
template <int CTRL>
DEV float dpp_swap(float x) {
  union { float f; int i; } a, r;
  a.f = x;
  r.i = __builtin_amdgcn_update_dpp(0, a.i, CTRL, 0xf, 0xf, true);
  return r.f;
}

// Reduce across each independent 16-lane half of the wave (rows of the WMMA C layout).
DEV float half16_max(float v) {
  v = fmaxf(v, dpp_swap<0xB1>(v));    // xor1
  v = fmaxf(v, dpp_swap<0x4E>(v));    // xor2
  v = fmaxf(v, dpp_swap<0x141>(v));   // xor4-equivalent (quads already uniform)
  v = fmaxf(v, dpp_swap<0x140>(v));   // xor8-equivalent (octets already uniform)
  return v;
}
DEV float half16_sum(float v) {
  v += dpp_swap<0xB1>(v);
  v += dpp_swap<0x4E>(v);
  v += dpp_swap<0x141>(v);
  v += dpp_swap<0x140>(v);
  return v;
}

static constexpr int NTOK  = 16384;  // tokens per batch (128 x 128)
static constexpr int CDIM  = 256;    // d_model
static constexpr int HWD   = 128;    // image side
static constexpr int DHEAD = 32;
static constexpr int GS    = 64;     // tokens per 8x8 window
static constexpr int NHEAD = 8;

__global__ void wconv_f32_to_f16(const float* __restrict__ w, _Float16* __restrict__ w16) {
  int i = blockIdx.x * blockDim.x + threadIdx.x;
  if (i < CDIM * CDIM) w16[i] = (_Float16)w[i];
}

__global__ __launch_bounds__(256)
void win_attn_fc(const float* __restrict__ q, const float* __restrict__ k,
                 const float* __restrict__ v, const _Float16* __restrict__ w16,
                 float* __restrict__ out)
{
  __shared__ _Float16 sQ [NHEAD][GS * DHEAD];   // [head][token][d]   32 KB (pre-scaled)
  __shared__ _Float16 sK [NHEAD][GS * DHEAD];   // [head][token][d]   32 KB
  __shared__ _Float16 sVT[NHEAD][DHEAD * GS];   // [head][d][token]   32 KB
  __shared__ _Float16 sP [NHEAD][16 * GS];      // P tile per head    16 KB
  __shared__ _Float16 sO [GS * CDIM];           // merged heads       32 KB

  const int b  = blockIdx.x >> 8;       // batch
  const int w  = blockIdx.x & 255;      // window (16x16 grid of 8x8 windows)
  const int wy = w >> 4, wx = w & 15;
  const int base_n = (wy * 8) * HWD + wx * 8;

  const int lane = threadIdx.x & 31;
  const int h    = threadIdx.x >> 5;    // wave id == head
  const int hi8  = (lane >> 4) * 8;     // upper half-wave K/M offset per WMMA layout
  const int l15  = lane & 15;

  const float scale = 0.17677669529663687f; // 1/sqrt(32), folded into Q at staging

  // ---- stage Q*scale, K (row-major) and V (transposed) as f16 into per-head LDS ----
  {
    const int tsub = lane >> 3;         // 4 tokens per iteration
    const int d4   = (lane & 7) * 4;    // 8 lanes cover d=0..31 as float4
    for (int it = 0; it < 16; ++it) {
      const int t = it * 4 + tsub;
      const int n = base_n + (t >> 3) * HWD + (t & 7);
      const long long g = ((long long)b * NTOK + n) * CDIM + h * DHEAD + d4;
      const float4 qv = *(const float4*)(q + g);
      const float4 kv = *(const float4*)(k + g);
      const float4 vv = *(const float4*)(v + g);
      H4 pq, pk;
      pq.h[0] = (_Float16)(qv.x * scale); pq.h[1] = (_Float16)(qv.y * scale);
      pq.h[2] = (_Float16)(qv.z * scale); pq.h[3] = (_Float16)(qv.w * scale);
      pk.h[0] = (_Float16)kv.x; pk.h[1] = (_Float16)kv.y;
      pk.h[2] = (_Float16)kv.z; pk.h[3] = (_Float16)kv.w;
      *(v2u*)&sQ[h][t * DHEAD + d4] = pq.u;
      *(v2u*)&sK[h][t * DHEAD + d4] = pk.u;
      sVT[h][(d4 + 0) * GS + t] = (_Float16)vv.x;
      sVT[h][(d4 + 1) * GS + t] = (_Float16)vv.y;
      sVT[h][(d4 + 2) * GS + t] = (_Float16)vv.z;
      sVT[h][(d4 + 3) * GS + t] = (_Float16)vv.w;
    }
  }
  // per-head LDS regions are written and read by the same wave; same-wave LDS
  // ops stay in order, compiler inserts s_wait_dscnt for data deps.

  for (int mi = 0; mi < 4; ++mi) {
    // A fragment: Q rows mi*16..mi*16+15, full K=32
    v16h qa = frag16(&sQ[h][(mi * 16 + l15) * DHEAD + hi8]);
    v8f acc[4];
    #pragma unroll
    for (int nt = 0; nt < 4; ++nt) {
      v16h kb = frag16(&sK[h][(nt * 16 + l15) * DHEAD + hi8]);  // B col = K token
      v8f z = {};
      acc[nt] = wmma_f16(qa, kb, z);
    }

    // ---- softmax over 64 cols; row r lives in VGPR r across one 16-lane half
    V8F s0, s1, s2, s3;
    s0.v = acc[0]; s1.v = acc[1]; s2.v = acc[2]; s3.v = acc[3];
    float rinv[8];
    #pragma unroll
    for (int r = 0; r < 8; ++r) {
      float mx = fmaxf(fmaxf(s0.f[r], s1.f[r]), fmaxf(s2.f[r], s3.f[r]));
      mx = half16_max(mx);
      float e0 = __expf(s0.f[r] - mx);
      float e1 = __expf(s1.f[r] - mx);
      float e2 = __expf(s2.f[r] - mx);
      float e3 = __expf(s3.f[r] - mx);
      s0.f[r] = e0; s1.f[r] = e1; s2.f[r] = e2; s3.f[r] = e3;
      rinv[r] = 1.0f / half16_sum(e0 + e1 + e2 + e3);
    }

    // C-layout P -> row-major 16x64 LDS tile (row = hi8+r, col = nt*16+l15)
    #pragma unroll
    for (int r = 0; r < 8; ++r) {
      sP[h][(hi8 + r) * GS +  0 + l15] = (_Float16)s0.f[r];
      sP[h][(hi8 + r) * GS + 16 + l15] = (_Float16)s1.f[r];
      sP[h][(hi8 + r) * GS + 32 + l15] = (_Float16)s2.f[r];
      sP[h][(hi8 + r) * GS + 48 + l15] = (_Float16)s3.f[r];
    }

    // ---- O = P @ V : K=64 in two 32-wide steps, N=32 in two 16-col tiles
    v16h pa0 = frag16(&sP[h][l15 * GS +  0 + hi8]);
    v16h pa1 = frag16(&sP[h][l15 * GS + 32 + hi8]);
    #pragma unroll
    for (int nt = 0; nt < 2; ++nt) {
      v16h vb0 = frag16(&sVT[h][(nt * 16 + l15) * GS +  0 + hi8]);
      v16h vb1 = frag16(&sVT[h][(nt * 16 + l15) * GS + 32 + hi8]);
      v8f z = {};
      v8f o = wmma_f16(pa0, vb0, z);
      o = wmma_f16(pa1, vb1, o);
      V8F ou; ou.v = o;
      #pragma unroll
      for (int r = 0; r < 8; ++r) {
        const int row = mi * 16 + hi8 + r;            // token in window
        const int col = h * DHEAD + nt * 16 + l15;    // merged channel
        sO[row * CDIM + col] = (_Float16)(ou.f[r] * rinv[r]);
      }
    }
  }

  __syncthreads();  // sO tile complete across all 8 head-waves

  // ---- FC: out_tile(64x256) = O(64x256) @ W^T ; B column n = W row n ----
  const int fmi = h & 3;    // 16-row M-tile
  const int fnh = h >> 2;   // 128-col N half
  v8f facc[8];
  {
    v8f z = {};
    #pragma unroll
    for (int i = 0; i < 8; ++i) facc[i] = z;
  }
  for (int ks = 0; ks < 8; ++ks) {
    v16h oa = frag16(&sO[(fmi * 16 + l15) * CDIM + ks * 32 + hi8]);
    #pragma unroll
    for (int nt = 0; nt < 8; ++nt) {
      const int ncol = fnh * 128 + nt * 16 + l15;
      v16h wb = frag16(&w16[ncol * CDIM + ks * 32 + hi8]);  // L2-resident
      facc[nt] = wmma_f16(oa, wb, facc[nt]);
    }
  }
  #pragma unroll
  for (int nt = 0; nt < 8; ++nt) {
    V8F t; t.v = facc[nt];
    #pragma unroll
    for (int r = 0; r < 8; ++r) {
      const int row = fmi * 16 + hi8 + r;                   // token in window
      const int n   = base_n + (row >> 3) * HWD + (row & 7);
      const int c   = fnh * 128 + nt * 16 + l15;
      out[((long long)b * NTOK + n) * CDIM + c] = t.f[r];
    }
  }
}

extern "C" void kernel_launch(void* const* d_in, const int* in_sizes, int n_in,
                              void* d_out, int out_size, void* d_ws, size_t ws_size,
                              hipStream_t stream) {
  const float* q = (const float*)d_in[0];
  const float* k = (const float*)d_in[1];
  const float* v = (const float*)d_in[2];
  const float* W = (const float*)d_in[3];
  float* out     = (float*)d_out;
  _Float16* w16  = (_Float16*)d_ws;   // 128 KB f16 copy of W_fc

  wconv_f32_to_f16<<<(CDIM * CDIM + 255) / 256, 256, 0, stream>>>(W, w16);
  win_attn_fc<<<8 * 256, 256, 0, stream>>>(q, k, v, w16, out);
}
